// GatedReadout_24747601560134
// MI455X (gfx1250) — matrix operand order repaired and checked by reading, not compiled, not run on hardware
//
#include <hip/hip_runtime.h>
#include <hip/hip_bf16.h>
#include <math.h>

// ---------------------------------------------------------------------------
// GatedReadout on MI455X (gfx1250, wave32, WMMA)
//   gate = sigmoid(nodes @ Wg + bg); feat = tanh(nodes @ Wf + bf)
//   gated = gate * feat * mask ; segment mean / max over sorted indicator
// v4: LDS bank-conflict-free operand layouts (row stride 264 bf16 -> each
//     ds_load_b128 hits 16 distinct bank quads), async-to-LDS B staging,
//     2 M-tiles per wave, WMMA bf16 dual-GEMM, fused segment reduction.
// ---------------------------------------------------------------------------

#define N_NODES   262144
#define D_IN      256
#define EMBED     256
#define NSEG      64
#define ROWS_WG   128           // rows per workgroup (8 M-tiles of 16)
#define KSTR      264           // padded bf16 row stride (132 dwords = 4 mod 64)

// workspace layout (bytes)
#define WS_SEGSUM 0             // f32 [64][256]
#define WS_SEGMAX 65536         // u32 [64][256] (order-preserving keys)
#define WS_SEGCNT 131072        // f32 [64]
#define WS_WT     131584        // bf16 [2][256 n][256 k]  (transposed weights)

typedef __bf16 v16bf __attribute__((ext_vector_type(16)));
typedef float  v8f   __attribute__((ext_vector_type(8)));
typedef int    v4i   __attribute__((vector_size(16)));   // matches async builtin

union V16BF { v16bf v; uint4 q[2]; };

#define TO_GLB_V4I(p) ((__attribute__((address_space(1))) v4i*)(p))
#define TO_LDS_V4I(p) ((__attribute__((address_space(3))) v4i*)(p))

#if __has_builtin(__builtin_amdgcn_global_load_async_to_lds_b128)
#define HAVE_ASYNC_LDS 1
#endif

__device__ __forceinline__ unsigned short f2bf(float x) {
    unsigned b = __float_as_uint(x);
    b += 0x7FFFu + ((b >> 16) & 1u);       // round to nearest even
    return (unsigned short)(b >> 16);
}

// monotone f32 -> u32 key so unsigned atomicMax == float max
__device__ __forceinline__ unsigned fkey(float x) {
    unsigned b = __float_as_uint(x);
    return (b & 0x80000000u) ? ~b : (b | 0x80000000u);
}

__device__ __forceinline__ void wait_async_lds() {
#if defined(HAVE_ASYNC_LDS)
#if __has_builtin(__builtin_amdgcn_s_wait_asynccnt)
    __builtin_amdgcn_s_wait_asynccnt(0);
#else
    asm volatile("s_wait_asynccnt 0x0" ::: "memory");
#endif
#endif
}

// ---------------------------------------------------------------------------
// Kernel 0: transpose+convert weights to bf16, zero segment accumulators.
// ---------------------------------------------------------------------------
__global__ __launch_bounds__(256) void gr_prep_kernel(
    const float* __restrict__ Wg, const float* __restrict__ Wf,
    char* __restrict__ ws)
{
    int t = blockIdx.x * blockDim.x + threadIdx.x;     // 0 .. 131071
    {
        int mat = t >> 16;
        int idx = t & 65535;
        int n = idx >> 8, k = idx & 255;
        const float* W = mat ? Wf : Wg;                 // W is [k][n] row-major
        ((unsigned short*)(ws + WS_WT))[mat * 65536 + n * 256 + k] =
            f2bf(W[k * EMBED + n]);
    }
    if (t < NSEG * EMBED) {
        ((float*)(ws + WS_SEGSUM))[t]    = 0.0f;
        ((unsigned*)(ws + WS_SEGMAX))[t] = 0u;          // < fkey(any real)
    }
    if (t < NSEG) ((float*)(ws + WS_SEGCNT))[t] = 0.0f;
}

// ---------------------------------------------------------------------------
// Kernel 1: fused dual-GEMM (WMMA bf16) + activation + segment reduction.
// 256 threads = 8 waves. wave w: mtile-pair p = w&3 (rows 32p..32p+31),
// ntset = w>>2 (8 N-tiles). Each B fragment feeds 4 WMMAs (2 mtiles x g/f).
// ---------------------------------------------------------------------------
__global__ __launch_bounds__(256) void gr_main_kernel(
    const float* __restrict__ nodes,
    const int*   __restrict__ indicator,
    const float* __restrict__ mask,
    const float* __restrict__ biasG,
    const float* __restrict__ biasF,
    char* __restrict__ ws)
{
    __shared__ __align__(16) unsigned short A_s[ROWS_WG][KSTR];   // ~66 KB bf16
    __shared__ __align__(16) unsigned short B_s[64 * KSTR];       // ~33 KB bf16
    __shared__ float G[2][ROWS_WG][17];                           // ~17 KB
    __shared__ int   seg_l[ROWS_WG];
    __shared__ float mask_l[ROWS_WG];

    float*    seg_sum = (float*)(ws + WS_SEGSUM);
    unsigned* seg_max = (unsigned*)(ws + WS_SEGMAX);
    float*    seg_cnt = (float*)(ws + WS_SEGCNT);
    const unsigned short* WT = (const unsigned short*)(ws + WS_WT);

    const int t = threadIdx.x;
    const int rowBase = blockIdx.x * ROWS_WG;

    // ---- stage A tile: global f32 -> LDS bf16 (each thread: 128 floats).
    // row = t&127, part = t>>7: conflicting store partners are in different
    // waves; within a wave rows are consecutive (2-way worst case).
    {
        int row = t & 127, part = t >> 7;
        const float4* src =
            (const float4*)(nodes + (size_t)(rowBase + row) * D_IN + part * 128);
#pragma unroll
        for (int j = 0; j < 32; ++j) {
            float4 v = src[j];
            unsigned lo = (unsigned)f2bf(v.x) | ((unsigned)f2bf(v.y) << 16);
            unsigned hi = (unsigned)f2bf(v.z) | ((unsigned)f2bf(v.w) << 16);
            *(uint2*)&A_s[row][part * 128 + j * 4] = make_uint2(lo, hi);
        }
    }
    if (t < ROWS_WG) {
        seg_l[t]  = indicator[rowBase + t];
        mask_l[t] = mask[rowBase + t];
    }

    // ---- stage B panels (async when available) ----
    // B_s: 64 logical rows (ntset,mat,n) x KSTR; real data = 256 bf16/row.
    // 2048 chunks of 16B; wave-consecutive chunks walk one row (coalesced).
    auto stage_b = [&](int i) {
#pragma unroll
        for (int j = 0; j < 8; ++j) {
            int c   = t + j * 256;                     // 0..2047
            int row = c >> 5;                          // (ntset*2+mat)*16 + n
            int k   = (c & 31) * 8;                    // ushort k offset
            int ntm = row >> 4, n = row & 15;
            int mat = ntm & 1, nts = ntm >> 1;
            int nt  = nts * 8 + i;
            const unsigned short* src = WT + mat * 65536 + (nt * 16 + n) * 256 + k;
            unsigned short* dst = &B_s[row * KSTR + k];
#if defined(HAVE_ASYNC_LDS)
            __builtin_amdgcn_global_load_async_to_lds_b128(
                TO_GLB_V4I(src), TO_LDS_V4I(dst), 0, 0);
#else
            *(uint4*)dst = *(const uint4*)src;
#endif
        }
    };
    stage_b(0);
    wait_async_lds();
    __syncthreads();

    const int wave  = t >> 5, lane = t & 31;
    const int p     = wave & 3;           // mtile pair: rows 32p..32p+31
    const int ntset = wave >> 2;          // 0..1
    const int laneLo = lane & 15, laneHi = lane >> 4;

    // ---- A fragments -> registers (contiguous k = 16*laneHi + s, same order as B), 2 M-tiles ----
    V16BF a[2][8];
#pragma unroll
    for (int mt = 0; mt < 2; ++mt) {
        int rowA = p * 32 + mt * 16 + laneLo;
#pragma unroll
        for (int kt = 0; kt < 8; ++kt) {
            int k0 = kt * 32 + laneHi * 16;            // K run {k0..+15}
            a[mt][kt].q[0] = *(const uint4*)&A_s[rowA][k0];
            a[mt][kt].q[1] = *(const uint4*)&A_s[rowA][k0 + 8];
        }
    }

    // ---- main loop over this wave's 8 N-tiles ----
    for (int i = 0; i < 8; ++i) {
        int nt = ntset * 8 + i;
        v8f cg0 = {0.f,0.f,0.f,0.f,0.f,0.f,0.f,0.f};
        v8f cf0 = {0.f,0.f,0.f,0.f,0.f,0.f,0.f,0.f};
        v8f cg1 = {0.f,0.f,0.f,0.f,0.f,0.f,0.f,0.f};
        v8f cf1 = {0.f,0.f,0.f,0.f,0.f,0.f,0.f,0.f};

        const unsigned short* pG = &B_s[((ntset * 2 + 0) * 16 + laneLo) * KSTR
                                        + laneHi * 16];
        const unsigned short* pF = &B_s[((ntset * 2 + 1) * 16 + laneLo) * KSTR
                                        + laneHi * 16];

#pragma unroll
        for (int kt = 0; kt < 8; ++kt) {
            V16BF bG, bF;
            bG.q[0] = *(const uint4*)(pG);
            bG.q[1] = *(const uint4*)(pG + 8);
            bF.q[0] = *(const uint4*)(pF);
            bF.q[1] = *(const uint4*)(pF + 8);
            cg0 = __builtin_amdgcn_wmma_f32_16x16x32_bf16(
                      false, a[0][kt].v, false, bG.v, (short)0, cg0, false, false);
            cf0 = __builtin_amdgcn_wmma_f32_16x16x32_bf16(
                      false, a[0][kt].v, false, bF.v, (short)0, cf0, false, false);
            cg1 = __builtin_amdgcn_wmma_f32_16x16x32_bf16(
                      false, a[1][kt].v, false, bG.v, (short)0, cg1, false, false);
            cf1 = __builtin_amdgcn_wmma_f32_16x16x32_bf16(
                      false, a[1][kt].v, false, bF.v, (short)0, cf1, false, false);
            pG += 32; pF += 32;
        }

        // ---- bias + activation + gated value -> LDS exchange ----
        int colg = nt * 16 + laneLo;
        float bGv = biasG[colg], bFv = biasF[colg];
#pragma unroll
        for (int mt = 0; mt < 2; ++mt) {
            const v8f& cg = mt ? cg1 : cg0;
            const v8f& cf = mt ? cf1 : cf0;
#pragma unroll
            for (int r = 0; r < 8; ++r) {
                int rloc = p * 32 + mt * 16 + laneHi * 8 + r;   // C: M = r + 8*hi
                float gate = 1.0f / (1.0f + __expf(-(cg[r] + bGv)));
                float e = __expf(2.0f * (cf[r] + bFv));
                float feat = (e - 1.0f) / (e + 1.0f);           // tanh
                G[ntset][rloc][laneLo] = gate * feat * mask_l[rloc];
            }
        }
        __syncthreads();   // G complete; everyone done reading B_s

        // ---- run-length segment reduction (indicator is sorted) ----
        {
            int slot  = t & 31;          // 32 columns this iteration (2 bufs x 16)
            int chunk = t >> 5;          // 8 row-chunks of 16 rows
            int buf   = slot >> 4;
            int c16   = slot & 15;
            int colOut = (buf * 8 + i) * 16 + c16;
            int r0 = chunk * 16;
            int cur = seg_l[r0];
            float s = 0.0f, mx = -3.402823466e38f;
#pragma unroll
            for (int r = 0; r < 16; ++r) {
                int row = r0 + r;
                int sg = seg_l[row];
                float v = G[buf][row][c16];
                if (sg != cur) {
                    atomicAdd(seg_sum + cur * EMBED + colOut, s);
                    atomicMax(seg_max + cur * EMBED + colOut, fkey(mx));
                    cur = sg; s = 0.0f; mx = -3.402823466e38f;
                }
                s += v; mx = fmaxf(mx, v);
            }
            atomicAdd(seg_sum + cur * EMBED + colOut, s);
            atomicMax(seg_max + cur * EMBED + colOut, fkey(mx));

            if (i == 0 && slot == 0) {   // per-segment mask counts, once per WG
                int cur2 = seg_l[r0]; float c = 0.0f;
#pragma unroll
                for (int r = 0; r < 16; ++r) {
                    int sg = seg_l[r0 + r];
                    if (sg != cur2) { atomicAdd(seg_cnt + cur2, c); cur2 = sg; c = 0.0f; }
                    c += mask_l[r0 + r];
                }
                atomicAdd(seg_cnt + cur2, c);
            }
        }

        if (i < 7) stage_b(i + 1);       // overlap staging with reduction tail
        wait_async_lds();
        __syncthreads();                 // B_s ready; G free to overwrite
    }
}

// ---------------------------------------------------------------------------
// Kernel 2: finalize -> out[64][512] = [mean | max]
// ---------------------------------------------------------------------------
__global__ __launch_bounds__(256) void gr_final_kernel(
    const char* __restrict__ ws, float* __restrict__ out)
{
    int t = blockIdx.x * blockDim.x + threadIdx.x;   // 0..16383
    int s = t >> 8, c = t & 255;
    const float*    seg_sum = (const float*)(ws + WS_SEGSUM);
    const unsigned* seg_max = (const unsigned*)(ws + WS_SEGMAX);
    const float*    seg_cnt = (const float*)(ws + WS_SEGCNT);
    float mean = seg_sum[t] / fmaxf(seg_cnt[s], 1e-6f);
    unsigned k = seg_max[t];
    unsigned b = (k & 0x80000000u) ? (k ^ 0x80000000u) : ~k;
    out[s * 512 + c]       = mean;
    out[s * 512 + 256 + c] = __uint_as_float(b);
}

extern "C" void kernel_launch(void* const* d_in, const int* in_sizes, int n_in,
                              void* d_out, int out_size, void* d_ws, size_t ws_size,
                              hipStream_t stream) {
    const float* nodes     = (const float*)d_in[0];
    const int*   indicator = (const int*)  d_in[1];
    const float* mask      = (const float*)d_in[2];
    const float* Wg        = (const float*)d_in[3];
    const float* biasG     = (const float*)d_in[4];
    const float* Wf        = (const float*)d_in[5];
    const float* biasF     = (const float*)d_in[6];
    char* ws  = (char*)d_ws;
    float* out = (float*)d_out;

    gr_prep_kernel<<<512, 256, 0, stream>>>(Wg, Wf, ws);
    gr_main_kernel<<<N_NODES / ROWS_WG, 256, 0, stream>>>(
        nodes, indicator, mask, biasG, biasF, ws);
    gr_final_kernel<<<NSEG, 256, 0, stream>>>(ws, out);
}
